// MultiHeadAttention_38517266710665
// MI455X (gfx1250) — matrix operand order repaired and checked by reading, not compiled, not run on hardware
//
#include <hip/hip_runtime.h>
#include <math.h>

typedef __bf16 bf16;
typedef bf16  v16bf __attribute__((ext_vector_type(16)));
typedef bf16  v8bf  __attribute__((ext_vector_type(8)));
typedef float v8f   __attribute__((ext_vector_type(8)));

#define WMMA_BF16(a, b, c) \
  __builtin_amdgcn_wmma_f32_16x16x32_bf16(false, (a), false, (b), (short)0, (c), false, false)

constexpr int B_ = 4, C_ = 768, T_ = 1024, H_ = 12, K_ = 64, NREL = 21;

// ---------------------------------------------------------------------------
// fp32 -> bf16 convert (grid-strided)
// ---------------------------------------------------------------------------
__global__ void f2bf_kernel(const float* __restrict__ src, bf16* __restrict__ dst, int n) {
  int i = blockIdx.x * blockDim.x + threadIdx.x;
  int stride = gridDim.x * blockDim.x;
  for (; i < n; i += stride) dst[i] = (bf16)src[i];
}

// bias table: tab[d] = log1p(d), d = |i-j| in [0, T)
__global__ void biastab_kernel(float* __restrict__ tab) {
  int t = blockIdx.x * blockDim.x + threadIdx.x;
  if (t < T_) tab[t] = log1pf((float)t);
}

// ---------------------------------------------------------------------------
// Input projections: out16x16 tile = W(16 x 768) x X(768 x 16) + bias
// MODE 0: q  -> [B,H,T,64] bf16, scaled by 1/sqrt(64)
// MODE 1: kT -> [B,H,64,T] bf16 (transposed so S-WMMA B-frag is contiguous)
// MODE 2: v  -> [B,H,T,64] bf16
// grid (T/16, C/16, B), block 32 (one wave per tile)
// ---------------------------------------------------------------------------
template <int MODE>
__global__ __launch_bounds__(32) void proj_kernel(const bf16* __restrict__ W,
                                                  const float* __restrict__ bias,
                                                  const bf16* __restrict__ X,
                                                  bf16* __restrict__ out) {
  const int lane = threadIdx.x & 31;
  const int lh = lane >> 4;        // lane-half select
  const int ln = lane & 15;
  const int t0 = blockIdx.x * 16;
  const int o0 = blockIdx.y * 16;
  const int b  = blockIdx.z;

  const bf16* xb = X + (size_t)b * C_ * T_;
  v8f acc = {};
  for (int kc = 0; kc < C_; kc += 32) {
    // A fragment: W row o0+ln, K-cols per CDNA5 16-bit A layout
    const bf16* wr = W + (size_t)(o0 + ln) * C_ + kc + lh * 8;
    v8bf alo = *(const v8bf*)(wr);
    v8bf ahi = *(const v8bf*)(wr + 16);
    v16bf a;
#pragma unroll
    for (int e = 0; e < 8; e++) { a[e] = alo[e]; a[8 + e] = ahi[e]; }
    // B fragment: column t0+ln of X, K-rows per CDNA5 16-bit layout (same as A)
    v16bf bfrag;
#pragma unroll
    for (int e = 0; e < 8; e++) {
      bfrag[e] = xb[(size_t)(kc + lh * 8 + e) * T_ + t0 + ln];
      bfrag[8 + e] = xb[(size_t)(kc + 16 + lh * 8 + e) * T_ + t0 + ln];
    }
    acc = WMMA_BF16(a, bfrag, acc);
  }

  const int h = o0 / K_;
  if constexpr (MODE == 1) {
    __shared__ float tile[16][17];
#pragma unroll
    for (int r = 0; r < 8; r++) tile[r + 8 * lh][ln] = acc[r] + bias[o0 + r + 8 * lh];
    __syncthreads();  // single-wave WG: lowers to S_NOP, LDS is wave-ordered
    if (lane < 16) {
      const int d = (o0 % K_) + lane;
      bf16* dst = out + ((size_t)(b * H_ + h) * K_ + d) * T_ + t0;
      v16bf row;
#pragma unroll
      for (int n = 0; n < 16; n++) row[n] = (bf16)tile[lane][n];
      *(v16bf*)dst = row;
    }
  } else {
    const float scale = (MODE == 0) ? 0.125f : 1.0f;  // 1/sqrt(K)
    const int t = t0 + ln;
    const int d0 = (o0 % K_) + 8 * lh;
    v8bf pkt;
#pragma unroll
    for (int r = 0; r < 8; r++)
      pkt[r] = (bf16)((acc[r] + bias[o0 + r + 8 * lh]) * scale);
    *(v8bf*)(out + ((size_t)(b * H_ + h) * T_ + t) * K_ + d0) = pkt;
  }
}

// ---------------------------------------------------------------------------
// band[bht][d] = sum_k qs[bht][k] * rel_k[d][k],  d in [0,21)  (delta = d-10)
// ---------------------------------------------------------------------------
__global__ void band_kernel(const bf16* __restrict__ q, const float* __restrict__ relk,
                            float* __restrict__ band) {
  int tid = blockIdx.x * blockDim.x + threadIdx.x;
  if (tid >= B_ * H_ * T_) return;
  const bf16* qr = q + (size_t)tid * K_;
  float qs[K_];
#pragma unroll
  for (int d = 0; d < K_; d++) qs[d] = (float)qr[d];
#pragma unroll 1
  for (int m = 0; m < NREL; m++) {
    const float* rk = relk + m * K_;
    float s = 0.f;
#pragma unroll
    for (int d = 0; d < K_; d++) s += qs[d] * rk[d];
    band[(size_t)tid * NREL + m] = s;
  }
}

// ---------------------------------------------------------------------------
// Flash attention: per wave = 16 Q rows; loop over 32-wide j chunks.
// S via 4 WMMAs, online softmax, P through LDS to A-layout, PV via 4 WMMAs.
// Saves O (unscaled-by-relv output, normalized) and per-row (m, l).
// grid (T/64, B*H), block 128 (4 independent waves)
// ---------------------------------------------------------------------------
__global__ __launch_bounds__(128) void flash_kernel(
    const bf16* __restrict__ q, const bf16* __restrict__ kT, const bf16* __restrict__ v,
    const float* __restrict__ band, const float* __restrict__ tab,
    float* __restrict__ O, float* __restrict__ ml) {
  __shared__ bf16 Plds[4][16][40];
  const int lane = threadIdx.x & 31;
  const int w = threadIdx.x >> 5;
  const int lh = lane >> 4, ln = lane & 15;
  const int bh = blockIdx.y;
  const int i0 = blockIdx.x * 64 + w * 16;

  const bf16* qb = q + (size_t)bh * T_ * K_;
  const bf16* kb = kT + (size_t)bh * K_ * T_;
  const bf16* vb = v + (size_t)bh * T_ * K_;
  const float* bb = band + (size_t)bh * T_ * NREL;

  // Q A-fragments for d in [0,32) and [32,64)
  const bf16* qr = qb + (size_t)(i0 + ln) * K_ + lh * 8;
  v16bf aq0, aq1;
  {
    v8bf x0 = *(const v8bf*)(qr);
    v8bf x1 = *(const v8bf*)(qr + 16);
    v8bf x2 = *(const v8bf*)(qr + 32);
    v8bf x3 = *(const v8bf*)(qr + 48);
#pragma unroll
    for (int e = 0; e < 8; e++) {
      aq0[e] = x0[e]; aq0[8 + e] = x1[e];
      aq1[e] = x2[e]; aq1[8 + e] = x3[e];
    }
  }

  float m[8], l[8];
#pragma unroll
  for (int r = 0; r < 8; r++) { m[r] = -3.0e38f; l[r] = 0.f; }
  v8f acc[4] = {};

  for (int jc = 0; jc < T_; jc += 32) {
    if (jc + 32 < T_) {
      __builtin_prefetch(kb + (size_t)lane * T_ + jc + 32, 0, 1);
      __builtin_prefetch(vb + (size_t)(jc + 32 + lane) * K_, 0, 1);
    }
    // ---- S = Q * K^T for two 16-col tiles
    v8f S[2];
#pragma unroll
    for (int n = 0; n < 2; n++) {
      const int j0 = jc + 16 * n;
      v16bf b0, b1;
#pragma unroll
      for (int e = 0; e < 8; e++) {
        b0[e] = kb[(size_t)(lh * 8 + e) * T_ + j0 + ln];
        b0[8 + e] = kb[(size_t)(16 + lh * 8 + e) * T_ + j0 + ln];
        b1[e] = kb[(size_t)(32 + lh * 8 + e) * T_ + j0 + ln];
        b1[8 + e] = kb[(size_t)(48 + lh * 8 + e) * T_ + j0 + ln];
      }
      v8f s = {};
      s = WMMA_BF16(aq0, b0, s);
      s = WMMA_BF16(aq1, b1, s);
      S[n] = s;
    }
    // ---- bias + band + online softmax (D-layout: row = r+8*lh, col = ln)
#pragma unroll
    for (int r = 0; r < 8; r++) {
      const int ig = i0 + r + 8 * lh;
      float s0 = S[0][r], s1 = S[1][r];
      const int d0 = (jc + ln) - ig;
      const int d1 = (jc + 16 + ln) - ig;
      const int a0 = d0 < 0 ? -d0 : d0;
      const int a1 = d1 < 0 ? -d1 : d1;
      s0 -= tab[a0];
      s1 -= tab[a1];
      if (d0 >= -10 && d0 <= 10) s0 += bb[(size_t)ig * NREL + d0 + 10];
      if (d1 >= -10 && d1 <= 10) s1 += bb[(size_t)ig * NREL + d1 + 10];
      float rm = fmaxf(s0, s1);
#pragma unroll
      for (int o = 1; o < 16; o <<= 1) rm = fmaxf(rm, __shfl_xor(rm, o, 16));
      const float mn = fmaxf(m[r], rm);
      const float sc = __expf(m[r] - mn);
      const float p0 = __expf(s0 - mn);
      const float p1 = __expf(s1 - mn);
      float rs = p0 + p1;
#pragma unroll
      for (int o = 1; o < 16; o <<= 1) rs += __shfl_xor(rs, o, 16);
      l[r] = l[r] * sc + rs;
      m[r] = mn;
#pragma unroll
      for (int kc = 0; kc < 4; kc++) acc[kc][r] *= sc;
      Plds[w][r + 8 * lh][ln] = (bf16)p0;
      Plds[w][r + 8 * lh][16 + ln] = (bf16)p1;
    }
    // ---- reload P as A fragment (16 x 32 over j)
    v16bf pA;
    {
      const bf16* pr = &Plds[w][ln][lh * 8];
#pragma unroll
      for (int e = 0; e < 8; e++) { pA[e] = pr[e]; pA[8 + e] = pr[16 + e]; }
    }
    // ---- O += P * V
#pragma unroll
    for (int kc = 0; kc < 4; kc++) {
      v16bf bv;
#pragma unroll
      for (int e = 0; e < 8; e++) {
        bv[e] = vb[(size_t)(jc + lh * 8 + e) * K_ + kc * 16 + ln];
        bv[8 + e] = vb[(size_t)(jc + 16 + lh * 8 + e) * K_ + kc * 16 + ln];
      }
      acc[kc] = WMMA_BF16(pA, bv, acc[kc]);
    }
  }

  // epilogue: normalize and store
#pragma unroll
  for (int r = 0; r < 8; r++) {
    const int ig = i0 + r + 8 * lh;
    const float inv = 1.0f / l[r];
#pragma unroll
    for (int kc = 0; kc < 4; kc++)
      O[((size_t)bh * T_ + ig) * K_ + kc * 16 + ln] = acc[kc][r] * inv;
    if (ln == 0) {
      ml[((size_t)bh * T_ + ig) * 2 + 0] = m[r];
      ml[((size_t)bh * T_ + ig) * 2 + 1] = l[r];
    }
  }
}

// ---------------------------------------------------------------------------
// rel-v post-pass: out[i] += sum_{|j-i|<=10} p[i,j]*rel_v[j-i+10]
// with p reconstructed from saved (m,l). Writes transposed [B,768,T] bf16.
// ---------------------------------------------------------------------------
__global__ __launch_bounds__(128) void relv_kernel(
    const bf16* __restrict__ q, const bf16* __restrict__ kT,
    const float* __restrict__ band, const float* __restrict__ tab,
    const float* __restrict__ ml, const float* __restrict__ O,
    const float* __restrict__ relv, bf16* __restrict__ attT) {
  int tid = blockIdx.x * blockDim.x + threadIdx.x;
  if (tid >= B_ * H_ * T_) return;
  const int bh = tid / T_;
  const int i = tid % T_;
  const int b = bh / H_, h = bh % H_;

  float qs[K_];
  const bf16* qr = q + (size_t)tid * K_;
#pragma unroll
  for (int d = 0; d < K_; d++) qs[d] = (float)qr[d];
  float out[K_];
  const float* Or = O + (size_t)tid * K_;
#pragma unroll
  for (int d = 0; d < K_; d++) out[d] = Or[d];
  const float mi = ml[(size_t)tid * 2 + 0];
  const float li = ml[(size_t)tid * 2 + 1];
  const bf16* kb = kT + (size_t)bh * K_ * T_;

#pragma unroll 1
  for (int dd = 0; dd < NREL; dd++) {
    const int j = i + dd - 10;
    if (j < 0 || j >= T_) continue;
    float dot = 0.f;
#pragma unroll
    for (int d = 0; d < K_; d++) dot += qs[d] * (float)kb[(size_t)d * T_ + j];
    const int ad = dd > 10 ? dd - 10 : 10 - dd;
    const float s = dot + band[(size_t)tid * NREL + dd] - tab[ad];
    const float p = __expf(s - mi) / li;
    const float* rv = relv + dd * K_;
#pragma unroll
    for (int d = 0; d < K_; d++) out[d] += p * rv[d];
  }
  bf16* dst = attT + ((size_t)b * C_ + h * K_) * T_ + i;
#pragma unroll
  for (int d = 0; d < K_; d++) dst[(size_t)d * T_] = (bf16)out[d];
}

// ---------------------------------------------------------------------------
// Output projection: out[b,o,t] = Wo x attT + bo (fp32, coalesced via LDS)
// grid (T/16, C/16, B), block 32
// ---------------------------------------------------------------------------
__global__ __launch_bounds__(32) void oproj_kernel(const bf16* __restrict__ Wo,
                                                   const float* __restrict__ bo,
                                                   const bf16* __restrict__ attT,
                                                   float* __restrict__ out) {
  __shared__ float tile[16][17];
  const int lane = threadIdx.x & 31;
  const int lh = lane >> 4, ln = lane & 15;
  const int t0 = blockIdx.x * 16;
  const int o0 = blockIdx.y * 16;
  const int b  = blockIdx.z;
  const bf16* xb = attT + (size_t)b * C_ * T_;

  v8f acc = {};
  for (int kc = 0; kc < C_; kc += 32) {
    const bf16* wr = Wo + (size_t)(o0 + ln) * C_ + kc + lh * 8;
    v8bf alo = *(const v8bf*)(wr);
    v8bf ahi = *(const v8bf*)(wr + 16);
    v16bf a;
#pragma unroll
    for (int e = 0; e < 8; e++) { a[e] = alo[e]; a[8 + e] = ahi[e]; }
    v16bf bfrag;
#pragma unroll
    for (int e = 0; e < 8; e++) {
      bfrag[e] = xb[(size_t)(kc + lh * 8 + e) * T_ + t0 + ln];
      bfrag[8 + e] = xb[(size_t)(kc + 16 + lh * 8 + e) * T_ + t0 + ln];
    }
    acc = WMMA_BF16(a, bfrag, acc);
  }
#pragma unroll
  for (int r = 0; r < 8; r++) tile[r + 8 * lh][ln] = acc[r] + bo[o0 + r + 8 * lh];
  __syncthreads();
  if (lane < 16) {
    float* dst = out + ((size_t)b * C_ + o0 + lane) * T_ + t0;
#pragma unroll
    for (int n = 0; n < 16; n++) dst[n] = tile[lane][n];
  }
}

// ---------------------------------------------------------------------------
extern "C" void kernel_launch(void* const* d_in, const int* in_sizes, int n_in,
                              void* d_out, int out_size, void* d_ws, size_t ws_size,
                              hipStream_t stream) {
  (void)in_sizes; (void)n_in; (void)out_size; (void)ws_size;
  const float* x  = (const float*)d_in[0];
  const float* c  = (const float*)d_in[1];
  const float* Wq = (const float*)d_in[2];
  const float* bq = (const float*)d_in[3];
  const float* Wk = (const float*)d_in[4];
  const float* bk = (const float*)d_in[5];
  const float* Wv = (const float*)d_in[6];
  const float* bv = (const float*)d_in[7];
  const float* Wo = (const float*)d_in[8];
  const float* bo = (const float*)d_in[9];
  const float* relk = (const float*)d_in[10];
  const float* relv = (const float*)d_in[11];
  float* out = (float*)d_out;

  char* ws = (char*)d_ws;
  size_t off = 0;
  auto alloc = [&](size_t bytes) -> void* {
    void* p = ws + off;
    off = (off + bytes + 255) & ~(size_t)255;
    return p;
  };
  const size_t nBCT = (size_t)B_ * C_ * T_;       // 3145728
  const size_t nW   = (size_t)C_ * C_;            // 589824
  const size_t nBHT = (size_t)B_ * H_ * T_;       // 49152

  bf16* xbf  = (bf16*)alloc(nBCT * 2);
  bf16* cbf  = (bf16*)alloc(nBCT * 2);
  bf16* wqb  = (bf16*)alloc(nW * 2);
  bf16* wkb  = (bf16*)alloc(nW * 2);
  bf16* wvb  = (bf16*)alloc(nW * 2);
  bf16* wob  = (bf16*)alloc(nW * 2);
  bf16* qbuf = (bf16*)alloc(nBHT * K_ * 2);
  bf16* kTb  = (bf16*)alloc(nBHT * K_ * 2);
  bf16* vbuf = (bf16*)alloc(nBHT * K_ * 2);
  float* band = (float*)alloc(nBHT * NREL * 4);
  float* ml   = (float*)alloc(nBHT * 2 * 4);
  float* O    = (float*)alloc(nBHT * K_ * 4);
  float* tab  = (float*)alloc(T_ * 4);
  bf16* attT  = xbf;  // reuse: xbf dead after q projection

  // 1) converts
  f2bf_kernel<<<4096, 256, 0, stream>>>(x, xbf, (int)nBCT);
  f2bf_kernel<<<4096, 256, 0, stream>>>(c, cbf, (int)nBCT);
  f2bf_kernel<<<1024, 256, 0, stream>>>(Wq, wqb, (int)nW);
  f2bf_kernel<<<1024, 256, 0, stream>>>(Wk, wkb, (int)nW);
  f2bf_kernel<<<1024, 256, 0, stream>>>(Wv, wvb, (int)nW);
  f2bf_kernel<<<1024, 256, 0, stream>>>(Wo, wob, (int)nW);
  biastab_kernel<<<4, 256, 0, stream>>>(tab);

  // 2) projections (WMMA)
  dim3 gp(T_ / 16, C_ / 16, B_);
  proj_kernel<0><<<gp, 32, 0, stream>>>(wqb, bq, xbf, qbuf);
  proj_kernel<1><<<gp, 32, 0, stream>>>(wkb, bk, cbf, kTb);
  proj_kernel<2><<<gp, 32, 0, stream>>>(wvb, bv, cbf, vbuf);

  // 3) relative-key band
  band_kernel<<<(int)((nBHT + 255) / 256), 256, 0, stream>>>(qbuf, relk, band);

  // 4) flash attention (WMMA)
  dim3 gf(T_ / 64, B_ * H_);
  flash_kernel<<<gf, 128, 0, stream>>>(qbuf, kTb, vbuf, band, tab, O, ml);

  // 5) relative-value post-pass + transpose to [B,768,T]
  relv_kernel<<<(int)((nBHT + 127) / 128), 128, 0, stream>>>(qbuf, kTb, band, tab, ml, O,
                                                             relv, attT);

  // 6) output projection (WMMA)
  oproj_kernel<<<gp, 32, 0, stream>>>(wob, bo, attT, out);
}
